// TR_22084721836073
// MI455X (gfx1250) — matrix operand order repaired and checked by reading, not compiled, not run on hardware
//
#include <hip/hip_runtime.h>
#include <hip/hip_bf16.h>
#include <math.h>

typedef __attribute__((ext_vector_type(16))) _Float16 v16h;
typedef __attribute__((ext_vector_type(8)))  _Float16 v8h;
typedef __attribute__((ext_vector_type(8)))  float    v8f;
typedef __attribute__((ext_vector_type(4)))  unsigned int v4u;
typedef __attribute__((ext_vector_type(8)))  int      v8i;
typedef __attribute__((ext_vector_type(4)))  int      v4i;

// Problem constants
#define NB     2
#define CDIM   256
#define HIMG   96
#define WIMG   96
#define HS     48           // subsampled h=w=hk=wk
#define SPA    (HS*HS)      // 2304 spatial positions
#define HEADS  8
#define DHEAD  32
#define PHALF  128          // POS_DIM/2
#define INV_S2 0.70710678118654752f

// LDS B-tile: 64 s-rows x 32 c-cols, padded 64B row -> 80B pitch (40 f16)
#define BPITCH 40

#if defined(__has_builtin)
#  if __has_builtin(__builtin_amdgcn_tensor_load_to_lds)
#    define HAVE_TDM 1
#  endif
#endif
#ifndef HAVE_TDM
#  define HAVE_TDM 0
#endif

// ---------------------------------------------------------------------------
// TDM: DMA a 64(s) x 32(c) f16 tile (row stride CDIM elems) from global into
// LDS, padding each 64B row by 16B -> 80B pitch. D# per CDNA5 ISA 8.3/8.4.
// ---------------------------------------------------------------------------
__device__ __forceinline__ void tdm_load_b_tile(unsigned int ldsOff,
                                                const _Float16* gptr) {
#if HAVE_TDM
    unsigned long long ga = (unsigned long long)(uintptr_t)gptr;
    v4u g0;
    g0[0] = 1u;                                    // count=1 (valid), no gather
    g0[1] = ldsOff;                                // lds_addr (bytes)
    g0[2] = (unsigned int)ga;                      // global_addr lo
    g0[3] = (unsigned int)(ga >> 32) | (2u << 30); // global_addr hi | type=2
    v8i g1;
    g1[0] = (int)((1u << 16)       // data_size = 1 -> 2 bytes
                | (1u << 20)       // pad_enable
                | (3u << 22)       // pad_interval code 3 -> every 16 DWORDs (64B)
                | (3u << 25));     // pad_amount  code 3 -> 4 DWORDs (16B)
    g1[1] = (int)(32u << 16);      // tensor_dim0[15:0]=32  (bits 79:48)
    g1[2] = (int)(64u << 16);      // tensor_dim0 hi=0 ; tensor_dim1[15:0]=64
    g1[3] = (int)(32u << 16);      // tensor_dim1 hi=0 ; tile_dim0=32
    g1[4] = 64;                    // tile_dim1=64 ; tile_dim2=0
    g1[5] = CDIM;                  // tensor_dim0_stride lo (elements)
    g1[6] = 0;
    g1[7] = 0;
    v4i z4 = {0, 0, 0, 0};
#  if defined(__clang_major__) && __clang_major__ >= 23
    v8i z8 = {0, 0, 0, 0, 0, 0, 0, 0};
    __builtin_amdgcn_tensor_load_to_lds(g0, g1, z4, z4, z8, 0);
#  else
    __builtin_amdgcn_tensor_load_to_lds(g0, g1, z4, z4, 0);
#  endif
    __builtin_amdgcn_s_wait_tensorcnt(0);
#else
    (void)ldsOff; (void)gptr;
#endif
}

// ---------------------------------------------------------------------------
// Elementwise / small kernels
// ---------------------------------------------------------------------------

__global__ void k_convert_f16(const float* __restrict__ src,
                              _Float16* __restrict__ dst, int count) {
    int i = blockIdx.x * blockDim.x + threadIdx.x;
    if (i < count) dst[i] = (_Float16)src[i];
}

// xs16[n][s][c] = x[n][c][2i][2j]   (N-major: c fastest)
__global__ void k_subsample(const float* __restrict__ x,
                            _Float16* __restrict__ xs16) {
    int idx = blockIdx.x * blockDim.x + threadIdx.x;
    if (idx >= NB * SPA * CDIM) return;
    int c = idx % CDIM;
    int s = (idx / CDIM) % SPA;
    int n = idx / (CDIM * SPA);
    int j = s % HS, i = s / HS;
    xs16[idx] = (_Float16)x[(((size_t)n * CDIM + c) * HIMG + 2 * i) * WIMG + 2 * j];
}

__global__ void k_emb(float* __restrict__ emb) {
    int idx = blockIdx.x * blockDim.x + threadIdx.x;
    if (idx >= HS * HS * 64) return;
    int f = idx & 63;
    int b = (idx >> 6) % HS;
    int a = idx / (64 * HS);
    float diff = (float)(a - b) * 2.0f;
    float dim  = __powf(1000.0f, (4.0f / 256.0f) * (float)f);
    float ang  = diff / dim;
    float* row = emb + ((size_t)a * HS + b) * PHALF;
    row[f]      = __sinf(ang);
    row[f + 64] = __cosf(ang);
}

__global__ void k_pf(const float* __restrict__ emb,
                     const float* __restrict__ Wgx,
                     const float* __restrict__ Wgy,
                     float* __restrict__ pfx, float* __restrict__ pfy) {
    int idx = blockIdx.x * blockDim.x + threadIdx.x;
    if (idx >= HEADS * HS * HS * DHEAD) return;
    int d = idx & 31;
    int b = (idx >> 5) % HS;
    int a = (idx / (32 * HS)) % HS;
    int i = idx / (32 * HS * HS);
    int o = i * DHEAD + d;
    const float* em = emb + ((size_t)a * HS + b) * PHALF;
    const float* gx = Wgx + (size_t)o * PHALF;
    const float* gy = Wgy + (size_t)o * PHALF;
    float sx = 0.f, sy = 0.f;
    for (int e = 0; e < PHALF; ++e) { sx += em[e] * gx[e]; sy += em[e] * gy[e]; }
    pfx[idx] = sx * INV_S2;
    pfy[idx] = sy * INV_S2;
}

__global__ void k_wkb(const float* __restrict__ ab, const float* __restrict__ Wk,
                      float* __restrict__ wkb) {
    int idx = blockIdx.x * blockDim.x + threadIdx.x;
    if (idx >= HEADS * CDIM) return;
    int c = idx & 255;
    int i = idx >> 8;
    float s = 0.f;
    for (int d = 0; d < DHEAD; ++d)
        s += ab[i * DHEAD + d] * Wk[(size_t)(i * DHEAD + d) * CDIM + c];
    wkb[idx] = s;
}

// ebias[n][i][kv] = sum_c wkb[i][c] * xs[n][kv][c]  (contiguous c now)
__global__ void k_ebias(const float* __restrict__ wkb,
                        const _Float16* __restrict__ xs16,
                        float* __restrict__ ebias) {
    int idx = blockIdx.x * blockDim.x + threadIdx.x;
    if (idx >= NB * HEADS * SPA) return;
    int s = idx % SPA;
    int i = (idx / SPA) % HEADS;
    int n = idx / (SPA * HEADS);
    const float* wb = wkb + (size_t)i * CDIM;
    const _Float16* xp = xs16 + ((size_t)n * SPA + s) * CDIM;
    float acc = 0.f;
    for (int c = 0; c < CDIM; ++c) acc += wb[c] * (float)xp[c];
    ebias[idx] = acc;
}

__global__ void k_exey(const float* __restrict__ q,
                       const float* __restrict__ pfx, const float* __restrict__ pfy,
                       float* __restrict__ ex, float* __restrict__ ey) {
    int idx = blockIdx.x * blockDim.x + threadIdx.x;
    if (idx >= NB * HEADS * SPA * HS) return;
    int t = idx % HS;
    int s = (idx / HS) % SPA;
    int i = (idx / (HS * SPA)) % HEADS;
    int n = idx / (HS * SPA * HEADS);
    int w = s % HS, h = s / HS;
    const float* qp = q + (((size_t)n * HEADS + i) * SPA + s) * DHEAD;
    const float* px = pfx + (((size_t)i * HS + w) * HS + t) * DHEAD;
    const float* py = pfy + (((size_t)i * HS + h) * HS + t) * DHEAD;
    float sx = 0.f, sy = 0.f;
    #pragma unroll
    for (int d = 0; d < DHEAD; ++d) { sx += qp[d] * px[d]; sy += qp[d] * py[d]; }
    ex[idx] = sx;
    ey[idx] = sy;
}

// ---------------------------------------------------------------------------
// WMMA GEMM v3: 4 waves/block, 64(M) x 64(N) tile, K=256 in steps of 32.
// B is N-major [n][s][c]; per k-step the 64x32 B tile is TDM-staged into LDS
// (padded 80B pitch); fragment reads are 2x ds_load_b128 per fragment.
// mode 0: q f32 [n][head][s][d]    mode 1: vT f16 [n][head][d][s]
// mode 2: proj f32 [n][o][s] + bias[o]
// ---------------------------------------------------------------------------
__global__ void k_gemm_wmma(const _Float16* __restrict__ A,
                            const _Float16* __restrict__ B,
                            float* __restrict__ outF32,
                            _Float16* __restrict__ outF16,
                            const float* __restrict__ bias,
                            int mode) {
    __shared__ _Float16 Bs[64 * BPITCH];        // 5120 B
    const int tid   = threadIdx.x;              // 0..127
    const int lane  = tid & 31;
    const int wave  = tid >> 5;
    const int obase = blockIdx.x * 64 + wave * 16;
    const int nbase = blockIdx.y * 64;
    const int n     = blockIdx.z;
    const int mrow  = lane & 15;
    const int hi    = lane >> 4;
    const _Float16* Bn = B + (size_t)n * SPA * CDIM;
    const unsigned int ldsOff = (unsigned int)(uintptr_t)&Bs[0];

    v8f acc[4] = {{}, {}, {}, {}};
    for (int kb = 0; kb < CDIM; kb += 32) {
        __syncthreads();                        // tile reuse fence
#if HAVE_TDM
        if (wave == 0)
            tdm_load_b_tile(ldsOff, Bn + (size_t)nbase * CDIM + kb);
#else
        {   // coalesced fallback: 64 rows x 64B, 2 threads/row x 32B
            int row = tid >> 1, seg = tid & 1;
            const v8h* src = (const v8h*)(Bn + (size_t)(nbase + row) * CDIM + kb + seg * 16);
            v8h a0 = src[0], a1 = src[1];
            *(v8h*)&Bs[row * BPITCH + seg * 16]     = a0;
            *(v8h*)&Bs[row * BPITCH + seg * 16 + 8] = a1;
        }
#endif
        __syncthreads();
        // A fragment: M=mrow ; K = kb + 16*hi + e (contiguous)
        const _Float16* ap = A + (size_t)(obase + mrow) * CDIM + kb + hi * 16;
        v8h alo = *(const v8h*)ap;
        v8h ahi = *(const v8h*)(ap + 8);
        v16h a = __builtin_shufflevector(alo, ahi, 0, 1, 2, 3, 4, 5, 6, 7,
                                                   8, 9, 10, 11, 12, 13, 14, 15);
        #pragma unroll
        for (int j = 0; j < 4; ++j) {
            // B fragment: N = j*16+mrow (LDS row), K = 16*hi + e (contiguous)
            const _Float16* bsp = &Bs[(j * 16 + mrow) * BPITCH + hi * 16];
            v8h blo = *(const v8h*)bsp;
            v8h bhi = *(const v8h*)(bsp + 8);
            v16h b = __builtin_shufflevector(blo, bhi, 0, 1, 2, 3, 4, 5, 6, 7,
                                                       8, 9, 10, 11, 12, 13, 14, 15);
            acc[j] = __builtin_amdgcn_wmma_f32_16x16x32_f16(false, a, false, b,
                                                            (short)0, acc[j],
                                                            false, false);
        }
    }
    #pragma unroll
    for (int j = 0; j < 4; ++j) {
        #pragma unroll
        for (int r = 0; r < 8; ++r) {
            int o = obase + r + (hi << 3);
            int s = nbase + j * 16 + mrow;
            float val = acc[j][r];
            if (mode == 0) {
                int head = o >> 5, d = o & 31;
                outF32[(((size_t)n * HEADS + head) * SPA + s) * DHEAD + d] = val;
            } else if (mode == 1) {
                int head = o >> 5, d = o & 31;
                outF16[(((size_t)n * HEADS + head) * DHEAD + d) * SPA + s] = (_Float16)val;
            } else {
                outF32[((size_t)n * CDIM + o) * SPA + s] = val + bias[o];
            }
        }
    }
}

// ---------------------------------------------------------------------------
// Flash attention: one wave = 16 queries x 2304 kv (72 chunks of 32).
// V transposed [n][head][d][s] -> B fragments are contiguous 128b loads.
// Output att16 is N-major [n][s][c] for the projection GEMM's B operand.
// ---------------------------------------------------------------------------
__global__ void k_flash_attn(const float* __restrict__ ex,     // [n][8][2304][48]
                             const float* __restrict__ ey,     // [n][8][2304][48]
                             const float* __restrict__ ebias,  // [n][8][2304]
                             const _Float16* __restrict__ vT,  // [n][8][32][2304]
                             _Float16* __restrict__ att16) {   // [n][2304][256]
    const int lane  = threadIdx.x;
    const int qbase = blockIdx.x * 16;
    const int head  = blockIdx.y;
    const int n     = blockIdx.z;
    const int mrow  = lane & 15;
    const int hi    = lane >> 4;

    const size_t nh = (size_t)n * HEADS + head;
    const float* exq = ex + (nh * SPA + qbase + mrow) * HS;
    const float* eyq = ey + (nh * SPA + qbase + mrow) * HS;
    const float* eb  = ebias + nh * SPA;
    const _Float16* v0base = vT + (nh * DHEAD + mrow) * SPA;        // d = mrow
    const _Float16* v1base = vT + (nh * DHEAD + mrow + 16) * SPA;   // d = mrow+16

    float mprev = -1e30f, lsum = 0.f;
    v8f acc0 = {}, acc1 = {};

    for (int kvb = 0; kvb < SPA; kvb += 32) {
        float E[16];
        float mloc = -1e30f;
        #pragma unroll
        for (int e = 0; e < 16; ++e) {
            int K  = (hi << 4) + e;    // A-frag K index
            int kk = kvb + K;
            int u  = kk / HS;
            int v  = kk - u * HS;
            float val = eyq[u] + exq[v] + eb[kk];
            E[e] = val;
            mloc = fmaxf(mloc, val);
        }
        mloc = fmaxf(mloc, __shfl_xor(mloc, 16, 32));
        float mnew = fmaxf(mprev, mloc);
        float psum = 0.f;
        v16h p;
        #pragma unroll
        for (int e = 0; e < 16; ++e) {
            float pe = __expf(E[e] - mnew);
            psum += pe;
            p[e] = (_Float16)pe;
        }
        psum += __shfl_xor(psum, 16, 32);
        float alpha = __expf(mprev - mnew);
        lsum  = lsum * alpha + psum;
        mprev = mnew;
        #pragma unroll
        for (int r = 0; r < 8; ++r) {
            float ar = __shfl(alpha, r + (hi << 3), 32);
            acc0[r] *= ar;
            acc1[r] *= ar;
        }
        const _Float16* p0 = v0base + kvb + hi * 16;
        const _Float16* p1 = v1base + kvb + hi * 16;
        v8h l0 = *(const v8h*)p0, h0 = *(const v8h*)(p0 + 8);
        v8h l1 = *(const v8h*)p1, h1 = *(const v8h*)(p1 + 8);
        v16h b0 = __builtin_shufflevector(l0, h0, 0, 1, 2, 3, 4, 5, 6, 7,
                                                  8, 9, 10, 11, 12, 13, 14, 15);
        v16h b1 = __builtin_shufflevector(l1, h1, 0, 1, 2, 3, 4, 5, 6, 7,
                                                  8, 9, 10, 11, 12, 13, 14, 15);
        acc0 = __builtin_amdgcn_wmma_f32_16x16x32_f16(false, p, false, b0,
                                                      (short)0, acc0, false, false);
        acc1 = __builtin_amdgcn_wmma_f32_16x16x32_f16(false, p, false, b1,
                                                      (short)0, acc1, false, false);
    }
    #pragma unroll
    for (int r = 0; r < 8; ++r) {
        int row = r + (hi << 3);
        float lr = __shfl(lsum, row, 32);
        float inv = 1.0f / lr;
        int s  = qbase + row;
        int c0 = head * DHEAD + mrow;
        att16[((size_t)n * SPA + s) * CDIM + c0]      = (_Float16)(acc0[r] * inv);
        att16[((size_t)n * SPA + s) * CDIM + c0 + 16] = (_Float16)(acc1[r] * inv);
    }
}

// Bilinear 48->96 (half-pixel) + residual
__global__ void k_resize_residual(const float* __restrict__ proj,
                                  const float* __restrict__ x,
                                  const float* __restrict__ gamma,
                                  float* __restrict__ out) {
    int idx = blockIdx.x * blockDim.x + threadIdx.x;
    if (idx >= NB * CDIM * HIMG * WIMG) return;
    int xc = idx % WIMG;
    int yc = (idx / WIMG) % HIMG;
    int c  = (idx / (WIMG * HIMG)) % CDIM;
    int n  = idx / (WIMG * HIMG * CDIM);
    float fy = ((float)yc + 0.5f) * 0.5f - 0.5f;
    float fx = ((float)xc + 0.5f) * 0.5f - 0.5f;
    int y0 = (int)floorf(fy), x0 = (int)floorf(fx);
    float wy = fy - (float)y0, wx = fx - (float)x0;
    int y0c = y0 < 0 ? 0 : y0, y1c = (y0 + 1) > (HS - 1) ? (HS - 1) : (y0 + 1);
    int x0c = x0 < 0 ? 0 : x0, x1c = (x0 + 1) > (HS - 1) ? (HS - 1) : (x0 + 1);
    const float* pn = proj + ((size_t)n * CDIM + c) * SPA;
    float v00 = pn[y0c * HS + x0c], v01 = pn[y0c * HS + x1c];
    float v10 = pn[y1c * HS + x0c], v11 = pn[y1c * HS + x1c];
    float bil = (1.f - wy) * ((1.f - wx) * v00 + wx * v01)
              +        wy  * ((1.f - wx) * v10 + wx * v11);
    out[idx] = gamma[0] * bil + x[idx];
}

// ---------------------------------------------------------------------------
// Host-side orchestration
// ---------------------------------------------------------------------------
static inline size_t alignUp(size_t v) { return (v + 255) & ~(size_t)255; }

extern "C" void kernel_launch(void* const* d_in, const int* in_sizes, int n_in,
                              void* d_out, int out_size, void* d_ws, size_t ws_size,
                              hipStream_t stream) {
    const float* x     = (const float*)d_in[0];
    const float* Wq    = (const float*)d_in[1];
    const float* Wk    = (const float*)d_in[2];
    const float* Wv    = (const float*)d_in[3];
    const float* Wgx   = (const float*)d_in[4];
    const float* Wgy   = (const float*)d_in[5];
    const float* ab    = (const float*)d_in[6];
    const float* Wproj = (const float*)d_in[7];
    const float* bproj = (const float*)d_in[8];
    const float* gamma = (const float*)d_in[9];
    float* out = (float*)d_out;

    char* p = (char*)d_ws;
    _Float16* xs16 = (_Float16*)p; p += alignUp((size_t)NB * SPA * CDIM * 2);
    _Float16* Wq16 = (_Float16*)p; p += alignUp((size_t)CDIM * CDIM * 2);
    _Float16* Wv16 = (_Float16*)p; p += alignUp((size_t)CDIM * CDIM * 2);
    _Float16* Wp16 = (_Float16*)p; p += alignUp((size_t)CDIM * CDIM * 2);
    float* emb   = (float*)p; p += alignUp((size_t)HS * HS * PHALF * 4);
    float* pfx   = (float*)p; p += alignUp((size_t)HEADS * HS * HS * DHEAD * 4);
    float* pfy   = (float*)p; p += alignUp((size_t)HEADS * HS * HS * DHEAD * 4);
    float* wkb   = (float*)p; p += alignUp((size_t)HEADS * CDIM * 4);
    float* ebias = (float*)p; p += alignUp((size_t)NB * HEADS * SPA * 4);
    float* qbuf  = (float*)p; p += alignUp((size_t)NB * HEADS * SPA * DHEAD * 4);
    _Float16* vT = (_Float16*)p; p += alignUp((size_t)NB * HEADS * SPA * DHEAD * 2);
    float* ex    = (float*)p; p += alignUp((size_t)NB * HEADS * SPA * HS * 4);
    float* ey    = (float*)p; p += alignUp((size_t)NB * HEADS * SPA * HS * 4);
    _Float16* att16 = (_Float16*)p; p += alignUp((size_t)NB * SPA * CDIM * 2);
    float* proj  = (float*)p; p += alignUp((size_t)NB * CDIM * SPA * 4);

    const int TB = 256;
    k_convert_f16<<<(CDIM * CDIM + TB - 1) / TB, TB, 0, stream>>>(Wq, Wq16, CDIM * CDIM);
    k_convert_f16<<<(CDIM * CDIM + TB - 1) / TB, TB, 0, stream>>>(Wv, Wv16, CDIM * CDIM);
    k_convert_f16<<<(CDIM * CDIM + TB - 1) / TB, TB, 0, stream>>>(Wproj, Wp16, CDIM * CDIM);
    k_subsample<<<(NB * SPA * CDIM + TB - 1) / TB, TB, 0, stream>>>(x, xs16);

    k_emb<<<(HS * HS * 64 + TB - 1) / TB, TB, 0, stream>>>(emb);
    k_pf<<<(HEADS * HS * HS * DHEAD + TB - 1) / TB, TB, 0, stream>>>(emb, Wgx, Wgy, pfx, pfy);
    k_wkb<<<(HEADS * CDIM + TB - 1) / TB, TB, 0, stream>>>(ab, Wk, wkb);
    k_ebias<<<(NB * HEADS * SPA + TB - 1) / TB, TB, 0, stream>>>(wkb, xs16, ebias);

    // q and v projections (WMMA, TDM/LDS-staged B)
    dim3 ggrid(CDIM / 64, SPA / 64, NB);
    k_gemm_wmma<<<ggrid, 128, 0, stream>>>(Wq16, xs16, qbuf, nullptr, nullptr, 0);
    k_gemm_wmma<<<ggrid, 128, 0, stream>>>(Wv16, xs16, nullptr, vT, nullptr, 1);

    k_exey<<<(NB * HEADS * SPA * HS + TB - 1) / TB, TB, 0, stream>>>(qbuf, pfx, pfy, ex, ey);

    dim3 fgrid(SPA / 16, HEADS, NB);
    k_flash_attn<<<fgrid, 32, 0, stream>>>(ex, ey, ebias, vT, att16);

    k_gemm_wmma<<<ggrid, 128, 0, stream>>>(Wp16, att16, proj, nullptr, bproj, 2);

    k_resize_residual<<<(NB * CDIM * HIMG * WIMG + TB - 1) / TB, TB, 0, stream>>>(
        proj, x, gamma, out);
}
